// RelPositionMultiHeadedAttention_27582279974926
// MI455X (gfx1250) — matrix operand order repaired and checked by reading, not compiled, not run on hardware
//
#include <hip/hip_runtime.h>
#include <hip/hip_bf16.h>

// ---- sizes (hardcoded from reference) ----
#define B_ 4
#define T_ 2048
#define C_ 512
#define H_ 8
#define D_ 64

typedef __bf16 bf16_t;
typedef __bf16 v16bf __attribute__((ext_vector_type(16)));
typedef __bf16 v8bf  __attribute__((ext_vector_type(8)));
typedef __bf16 v4bf  __attribute__((ext_vector_type(4)));
typedef float  v8f   __attribute__((ext_vector_type(8)));

// native f32 -> bf16 (round-to-nearest-even fptrunc; gfx1250 has bf16 cvt)
__device__ __forceinline__ bf16_t f2bf(float f) { return (bf16_t)f; }

__device__ __forceinline__ v16bf cat8(v8bf lo, v8bf hi) {
    return __builtin_shufflevector(lo, hi, 0, 1, 2, 3, 4, 5, 6, 7,
                                           8, 9, 10, 11, 12, 13, 14, 15);
}

__device__ __forceinline__ v8f wmma_bf16(v16bf a, v16bf b, v8f c) {
    // D(f32,16x16) = A(bf16,16x32) * B(bf16,32x16) + C
    return __builtin_amdgcn_wmma_f32_16x16x32_bf16(false, a, false, b,
                                                   (short)0, c, false, false);
}

// A-fragment (16x32 bf16) from a row-major matrix with row stride `ld`.
// lane: M = lane&15 (row given by caller), group g = lane>>4.
// element e: K = (e<8?0:16) + g*8 + (e&7)  -> two contiguous 16B chunks.
__device__ __forceinline__ v16bf load_a_frag(const bf16_t* __restrict__ mat,
                                             int row, int ld, int d0, int g) {
    const bf16_t* p0 = mat + (unsigned)row * ld + d0 + g * 8;
    return cat8(*(const v8bf*)p0, *(const v8bf*)(p0 + 16));
}

// B-fragment (32x16 bf16), B[k,n] = mat[col_row, k] with row stride `ld`.
// lane: N = lane&15 (col_row given by caller), g = lane>>4 -> K = (e<8?0:16) + g*8 + (e&7),
// two contiguous 16B chunks per lane.
__device__ __forceinline__ v16bf load_b_frag(const bf16_t* __restrict__ mat,
                                             int col_row, int ld, int k0, int g) {
    const bf16_t* p0 = mat + (unsigned)col_row * ld + k0 + g * 8;
    return cat8(*(const v8bf*)p0, *(const v8bf*)(p0 + 16));
}

// ---------------------------------------------------------------------------
// GEMM: Y[8192,512] = A[8192,512] @ W[512,512] (+bias), bf16 WMMA, f32 acc.
// MODE 0: out0 = bf16(Y) reshaped [B,H,T,D]          (k / p projections)
// MODE 1: out0/1 = bf16((Y + pbu/pbv) * qscale)      (q proj, log2-domain)
// MODE 2: outf = Y (fp32, [B*T, C])                  (output projection)
// MODE 3: out0 = bf16(Y) transposed  [B,H,D,T]       (v projection)
// ---------------------------------------------------------------------------
template<int MODE, bool ABF16, bool HAS_BIAS>
__global__ __launch_bounds__(128) void gemm_kernel(
    const void* __restrict__ Ain,
    const float* __restrict__ W, const float* __restrict__ bias,
    bf16_t* __restrict__ out0, bf16_t* __restrict__ out1,
    const float* __restrict__ pbu, const float* __restrict__ pbv,
    float* __restrict__ outf)
{
    __shared__ bf16_t Abuf[64][32];   // [m][k]
    __shared__ bf16_t Bbuf[64][32];   // [n][k]  (transposed W tile)

    const int t    = threadIdx.x;
    const int lane = t & 31;
    const int w    = t >> 5;           // wave id 0..3, owns rows 16w..16w+15
    const int m0   = blockIdx.x * 64;
    const int n0   = blockIdx.y * 64;
    const int mloc = lane & 15;
    const int g    = lane >> 4;

    v8f acc[4];
#pragma unroll
    for (int nt = 0; nt < 4; ++nt)
#pragma unroll
        for (int v = 0; v < 8; ++v) acc[nt][v] = 0.f;

    for (int k0 = 0; k0 < C_; k0 += 32) {
        // ---- stage A tile 64x32 -> bf16 LDS (row-major) ----
        if (ABF16) {
            const bf16_t* Abf = (const bf16_t*)Ain;
            // pure copy: use CDNA5 async global->LDS DMA (16B per lane)
#pragma unroll
            for (int p = 0; p < 2; ++p) {
                int chunk = t + p * 128;            // 256 chunks of 8 bf16
                int r = chunk >> 2;
                int c = (chunk & 3) * 8;
                unsigned lds = (unsigned)(uintptr_t)&Abuf[r][c];
                const bf16_t* gp = Abf + (unsigned)(m0 + r) * C_ + k0 + c;
                asm volatile("global_load_async_to_lds_b128 %0, %1, off"
                             :: "v"(lds), "v"(gp) : "memory");
            }
        } else {
            const float* Af = (const float*)Ain;
#pragma unroll
            for (int p = 0; p < 4; ++p) {
                int chunk = t + p * 128;            // 512 chunks of 4 floats
                int r = chunk >> 3;
                int c = (chunk & 7) * 4;
                float4 d = *(const float4*)(Af + (unsigned)(m0 + r) * C_ + k0 + c);
                v4bf pk;
                pk[0] = f2bf(d.x); pk[1] = f2bf(d.y);
                pk[2] = f2bf(d.z); pk[3] = f2bf(d.w);
                *(v4bf*)&Abuf[r][c] = pk;
            }
        }
        // ---- stage W tile 32x64 -> bf16 LDS transposed: Bbuf[n][k] ----
#pragma unroll
        for (int p = 0; p < 4; ++p) {
            int chunk = t + p * 128;                // 512 chunks of 4 floats
            int r = chunk >> 4;                     // k-local
            int c = (chunk & 15) * 4;               // n-local
            float4 d = *(const float4*)(W + (unsigned)(k0 + r) * C_ + n0 + c);
            Bbuf[c + 0][r] = f2bf(d.x);
            Bbuf[c + 1][r] = f2bf(d.y);
            Bbuf[c + 2][r] = f2bf(d.z);
            Bbuf[c + 3][r] = f2bf(d.w);
        }
        if (ABF16) asm volatile("s_wait_asynccnt 0x0" ::: "memory");
        __syncthreads();

        // A fragment: two contiguous 16B LDS loads
        v16bf a = cat8(*(const v8bf*)&Abuf[16 * w + mloc][g * 8],
                       *(const v8bf*)&Abuf[16 * w + mloc][16 + g * 8]);
#pragma unroll
        for (int nt = 0; nt < 4; ++nt) {
            // B fragment: two contiguous 16B chunks along k in Bbuf[n][k]
            const bf16_t* bp = &Bbuf[nt * 16 + mloc][g * 8];
            v16bf bfr = cat8(*(const v8bf*)bp, *(const v8bf*)(bp + 16));
            acc[nt] = wmma_bf16(a, bfr, acc[nt]);
        }
        __syncthreads();
    }

    // log2-domain score scaling folded into q: (1/sqrt(D)) * log2(e)
    const float qscale = 0.125f * 1.44269504088896f;

    // ---- epilogue (fully resolved at compile time) ----
#pragma unroll
    for (int nt = 0; nt < 4; ++nt) {
#pragma unroll
        for (int v = 0; v < 8; ++v) {
            int mm  = v + 8 * g;
            int row = m0 + 16 * w + mm;       // [0, B*T)
            int n   = n0 + nt * 16 + mloc;    // [0, C)
            float val = acc[nt][v];
            if (HAS_BIAS) val += bias[n];
            if constexpr (MODE == 2) {
                outf[(unsigned)row * C_ + n] = val;
            } else {
                int bb = row >> 11;            // /T_
                int tt = row & (T_ - 1);
                int hh = n >> 6;
                int dd = n & 63;
                if constexpr (MODE == 3) {     // V^T : [B,H,D,T]
                    out0[((unsigned)(bb * H_ + hh) * D_ + dd) * T_ + tt] = f2bf(val);
                } else if constexpr (MODE == 1) {
                    unsigned o = ((unsigned)(bb * H_ + hh) * T_ + tt) * D_ + dd;
                    out0[o] = f2bf((val + pbu[hh * D_ + dd]) * qscale);
                    out1[o] = f2bf((val + pbv[hh * D_ + dd]) * qscale);
                } else {
                    out0[((unsigned)(bb * H_ + hh) * T_ + tt) * D_ + dd] = f2bf(val);
                }
            }
        }
    }
}

// ---------------------------------------------------------------------------
// Flash attention with Transformer-XL relative shift (scores in log2 domain).
// One wave (32 threads) per (b, h, 16-query-row strip). Streams 32 keys/step.
//   shift: out[i,j] = P[i, T-1-i+j]      (j <= i)
//                   = 0                  (j == i+1)
//                   = P[i+1, j-i-2]      (j >  i+1)
// Region-1 P tiles: rows r0..r0+15, window base b1 = T-16-r0+j0s (31 cols)
// Region-2 P tiles: rows r0+1..r0+16, window base b2 = j0s-r0-17 (31 cols)
// Both gather at in-window offset 15 - m + dj.
// l is kept lane-distributed and reduced once after the stream loop.
// ---------------------------------------------------------------------------
__global__ __launch_bounds__(32) void attn_kernel(
    const bf16_t* __restrict__ qu, const bf16_t* __restrict__ qv,
    const bf16_t* __restrict__ kk, const bf16_t* __restrict__ pp,
    const bf16_t* __restrict__ vt, bf16_t* __restrict__ ctx)
{
    __shared__ float  P1[16][32];
    __shared__ float  P2[16][32];
    __shared__ bf16_t Sb[16][32];

    const int lane = threadIdx.x;
    const int r0   = blockIdx.x * 16;
    const int h    = blockIdx.y;
    const int b    = blockIdx.z;
    const unsigned off = (unsigned)(b * H_ + h) * T_ * D_;
    qu += off; qv += off; kk += off; pp += off; vt += off;

    const int m = lane & 15;   // frag column / A-row
    const int g = lane >> 4;   // K-group / row half

    // A-fragments: qu rows r0.., qv rows r0.., qv rows r0+1.. (clamped)
    const int rowq  = r0 + m;
    const int rowq1 = (r0 + 1 + m < T_) ? (r0 + 1 + m) : (T_ - 1);
    v16bf quA[2], qvA[2], qv1A[2];
#pragma unroll
    for (int kc = 0; kc < 2; ++kc) {
        quA [kc] = load_a_frag(qu, rowq,  D_, 32 * kc, g);
        qvA [kc] = load_a_frag(qv, rowq,  D_, 32 * kc, g);
        qv1A[kc] = load_a_frag(qv, rowq1, D_, 32 * kc, g);
    }

    float mrow[8], lsum[8];
    v8f O[4];
#pragma unroll
    for (int v = 0; v < 8; ++v) { mrow[v] = -1e30f; lsum[v] = 0.f; }
#pragma unroll
    for (int dt = 0; dt < 4; ++dt)
#pragma unroll
        for (int v = 0; v < 8; ++v) O[dt][v] = 0.f;

    for (int j0 = 0; j0 < T_; j0 += 32) {
        float pvals[2][8];
#pragma unroll
        for (int s = 0; s < 2; ++s) {
            const int j0s = j0 + 16 * s;
            // content scores: (q+u) @ K^T   (already scaled, log2 domain)
            v8f acc;
#pragma unroll
            for (int v = 0; v < 8; ++v) acc[v] = 0.f;
            {
                v16bf kb0 = load_b_frag(kk, j0s + m, D_, 0, g);
                v16bf kb1 = load_b_frag(kk, j0s + m, D_, 32, g);
                acc = wmma_bf16(quA[0], kb0, acc);
                acc = wmma_bf16(quA[1], kb1, acc);
            }
            const bool reg1 = (j0s <= r0 + 15);      // any j <= i in tile
            const bool reg2 = (j0s + 15 >= r0 + 2);  // any j >  i+1 in tile
            if (reg1) {
                const int b1 = T_ - 16 - r0 + j0s;
#pragma unroll
                for (int t2 = 0; t2 < 2; ++t2) {
                    int cr = b1 + 16 * t2 + m;
                    cr = cr < 0 ? 0 : (cr > T_ - 1 ? T_ - 1 : cr);
                    v16bf pb0 = load_b_frag(pp, cr, D_, 0, g);
                    v16bf pb1 = load_b_frag(pp, cr, D_, 32, g);
                    v8f pt;
#pragma unroll
                    for (int v = 0; v < 8; ++v) pt[v] = 0.f;
                    pt = wmma_bf16(qvA[0], pb0, pt);
                    pt = wmma_bf16(qvA[1], pb1, pt);
#pragma unroll
                    for (int v = 0; v < 8; ++v) P1[v + 8 * g][16 * t2 + m] = pt[v];
                }
            }
            if (reg2) {
                const int b2 = j0s - r0 - 17;
#pragma unroll
                for (int t2 = 0; t2 < 2; ++t2) {
                    int cr = b2 + 16 * t2 + m;
                    cr = cr < 0 ? 0 : (cr > T_ - 1 ? T_ - 1 : cr);
                    v16bf pb0 = load_b_frag(pp, cr, D_, 0, g);
                    v16bf pb1 = load_b_frag(pp, cr, D_, 32, g);
                    v8f pt;
#pragma unroll
                    for (int v = 0; v < 8; ++v) pt[v] = 0.f;
                    pt = wmma_bf16(qv1A[0], pb0, pt);
                    pt = wmma_bf16(qv1A[1], pb1, pt);
#pragma unroll
                    for (int v = 0; v < 8; ++v) P2[v + 8 * g][16 * t2 + m] = pt[v];
                }
            }
            __syncthreads();
            // gather shifted pos (diagonal gather -> scalar LDS reads)
#pragma unroll
            for (int v = 0; v < 8; ++v) {
                int mm = v + 8 * g;        // frag row
                int i  = r0 + mm;
                int j  = j0s + m;
                int cidx = 15 - mm + m;    // in [0,30] for both regions
                float p1v = P1[mm][cidx];
                float p2v = P2[mm][cidx];
                float pos = (j <= i) ? p1v : ((j == i + 1) ? 0.f : p2v);
                pvals[s][v] = acc[v] + pos;
            }
            __syncthreads();
        }

        // online softmax (log2 domain); only the row-max needs a reduction,
        // l stays lane-distributed (reduced once after the loop)
        float alpha[8];
#pragma unroll
        for (int v = 0; v < 8; ++v) {
            float tmax = fmaxf(pvals[0][v], pvals[1][v]);
#pragma unroll
            for (int msk = 1; msk < 16; msk <<= 1)
                tmax = fmaxf(tmax, __shfl_xor(tmax, msk, 32));
            float mnew = fmaxf(mrow[v], tmax);
            float a  = exp2f(mrow[v] - mnew);
            float p0 = exp2f(pvals[0][v] - mnew);
            float p1 = exp2f(pvals[1][v] - mnew);
            pvals[0][v] = p0; pvals[1][v] = p1;
            lsum[v] = lsum[v] * a + (p0 + p1);
            mrow[v] = mnew;
            alpha[v] = a;
        }
#pragma unroll
        for (int dt = 0; dt < 4; ++dt)
#pragma unroll
            for (int v = 0; v < 8; ++v) O[dt][v] *= alpha[v];

        // probs -> LDS as bf16 (A-layout-friendly: row-major 16x32)
#pragma unroll
        for (int s = 0; s < 2; ++s)
#pragma unroll
            for (int v = 0; v < 8; ++v) Sb[v + 8 * g][16 * s + m] = f2bf(pvals[s][v]);
        __syncthreads();
        // prob A-fragment: two contiguous 16B LDS loads
        v16bf pa = cat8(*(const v8bf*)&Sb[m][g * 8],
                        *(const v8bf*)&Sb[m][16 + g * 8]);
        // O += P @ V : B-fragment straight from global V^T (two 16B chunks per lane)
#pragma unroll
        for (int dt = 0; dt < 4; ++dt) {
            v16bf vb = load_b_frag(vt, dt * 16 + m, T_, j0, g);
            O[dt] = wmma_bf16(pa, vb, O[dt]);
        }
        __syncthreads();
    }

    // final l reduction (once) + normalize + store ctx bf16 [B*T, C]
#pragma unroll
    for (int v = 0; v < 8; ++v) {
        float ls = lsum[v];
#pragma unroll
        for (int msk = 1; msk < 16; msk <<= 1)
            ls += __shfl_xor(ls, msk, 32);
        float inv = 1.0f / ls;
        int mm = v + 8 * g;
        int trow = r0 + mm;
#pragma unroll
        for (int dt = 0; dt < 4; ++dt) {
            float val = O[dt][v] * inv;
            ctx[(unsigned)(b * T_ + trow) * C_ + h * D_ + dt * 16 + m] = f2bf(val);
        }
    }
}

// ---------------------------------------------------------------------------
extern "C" void kernel_launch(void* const* d_in, const int* in_sizes, int n_in,
                              void* d_out, int out_size, void* d_ws, size_t ws_size,
                              hipStream_t stream)
{
    (void)in_sizes; (void)n_in; (void)out_size; (void)ws_size;
    const float* query = (const float*)d_in[0];
    const float* key_  = (const float*)d_in[1];
    const float* value = (const float*)d_in[2];
    const float* posE  = (const float*)d_in[3];
    const float* Wq    = (const float*)d_in[4];
    const float* bq    = (const float*)d_in[5];
    const float* Wk    = (const float*)d_in[6];
    const float* bk    = (const float*)d_in[7];
    const float* Wv    = (const float*)d_in[8];
    const float* bv    = (const float*)d_in[9];
    const float* Wpos  = (const float*)d_in[10];
    const float* pbu   = (const float*)d_in[11];
    const float* pbv   = (const float*)d_in[12];
    const float* Wo    = (const float*)d_in[13];
    const float* bo    = (const float*)d_in[14];

    const size_t N = (size_t)B_ * T_ * C_;
    bf16_t* quB  = (bf16_t*)d_ws;      // (q + u) * qscale  [B,H,T,D]
    bf16_t* qvB  = quB + N;            // (q + v) * qscale  [B,H,T,D]
    bf16_t* kB   = qvB + N;            // k                 [B,H,T,D]
    bf16_t* pB   = kB  + N;            // p                 [B,H,T,D]
    bf16_t* vtB  = pB  + N;            // v transposed      [B,H,D,T]
    bf16_t* ctxB = vtB + N;            // attention output  [B*T, C]

    dim3 ggrid(B_ * T_ / 64, C_ / 64);
    // q projection -> two biased, pre-scaled bf16 copies
    gemm_kernel<1, false, true><<<ggrid, 128, 0, stream>>>(
        query, Wq, bq, quB, qvB, pbu, pbv, nullptr);
    gemm_kernel<0, false, true><<<ggrid, 128, 0, stream>>>(
        key_, Wk, bk, kB, nullptr, nullptr, nullptr, nullptr);
    gemm_kernel<3, false, true><<<ggrid, 128, 0, stream>>>(
        value, Wv, bv, vtB, nullptr, nullptr, nullptr, nullptr);
    gemm_kernel<0, false, false><<<ggrid, 128, 0, stream>>>(
        posE, Wpos, nullptr, pB, nullptr, nullptr, nullptr, nullptr);

    dim3 agrid(T_ / 16, H_, B_);
    attn_kernel<<<agrid, 32, 0, stream>>>(quB, qvB, kB, pB, vtB, ctxB);

    // output projection (fp32 result straight into d_out, A = bf16 ctx via
    // async global->LDS staging)
    gemm_kernel<2, true, true><<<ggrid, 128, 0, stream>>>(
        ctxB, Wo, bo, nullptr, nullptr, nullptr, nullptr, (float*)d_out);
}
